// TransformerEncoderLayer_20667382628721
// MI455X (gfx1250) — compile-verified, matrix operand order checked by reading, not run on hardware
//
#include <hip/hip_runtime.h>
#include <hip/hip_bf16.h>

// MI455X / gfx1250, wave32. WMMA f32_16x16x32_f16 for every matmul,
// GLOBAL_LOAD_ASYNC_TO_LDS_B128 (ASYNCcnt) for double-buffered tile staging.

typedef _Float16 f16;
typedef _Float16 v8h  __attribute__((ext_vector_type(8)));
typedef _Float16 v16h __attribute__((ext_vector_type(16)));
typedef float    v8f  __attribute__((ext_vector_type(8)));
typedef int      b128_t __attribute__((vector_size(16)));  // async builtin pointee

#define D_MODEL 1024
#define N_HEADS 16
#define DH      64
#define DIM_FF  4096
#define BSZ     4
#define SEQ     1024
#define MTOK    (BSZ * SEQ)

// ---------------- CDNA5 async copy (global -> LDS), with sync fallback -----
#if defined(__has_builtin)
#if __has_builtin(__builtin_amdgcn_global_load_async_to_lds_b128)
#define USE_ASYNC_LDS 1
#endif
#endif
#ifndef USE_ASYNC_LDS
#define USE_ASYNC_LDS 0
#endif

__device__ inline void async_cp16(const void* g, void* l) {
#if USE_ASYNC_LDS
  // Flat LDS addresses are {aperture, lds_offset[31:0]}; low 32 bits are the
  // DS offset, so an int->addrspace(3) cast builds a valid LDS pointer.
  __builtin_amdgcn_global_load_async_to_lds_b128(
      (__attribute__((address_space(1))) b128_t*)(unsigned long long)(uintptr_t)g,
      (__attribute__((address_space(3))) b128_t*)(unsigned)(uintptr_t)l, 0, 0);
#else
  *(v8h*)l = *(const v8h*)g;
#endif
}

#if USE_ASYNC_LDS
#if __has_builtin(__builtin_amdgcn_s_wait_asynccnt)
#define WAIT_ASYNC(n) __builtin_amdgcn_s_wait_asynccnt(n)
#else
#define WAIT_ASYNC(n) asm volatile("s_wait_asynccnt %0" ::"n"(n) : "memory")
#endif
#else
#define WAIT_ASYNC(n) ((void)0)
#endif

// ---------------------------------------------------------------- helpers
__device__ inline v8f vzero8() {
  v8f z;
#pragma unroll
  for (int e = 0; e < 8; ++e) z[e] = 0.0f;
  return z;
}

__device__ inline v8f wmma16(v16h a, v16h b, v8f c) {
  return __builtin_amdgcn_wmma_f32_16x16x32_f16(false, a, false, b, (short)0, c,
                                                false, false);
}

// 16x32 f16 operand fragment (A or B) from a row-major buffer.
// lane m = lane&15; lanes 0-15 K-base 0, lanes 16-31 K-base 8;
// VGPR0..3 = K kb..kb+7, VGPR4..7 = K kb+16..kb+23.
__device__ inline v16h ld_frag(const f16* rowbase, int stride, int lane) {
  int r  = lane & 15;
  int kb = (lane & 16) ? 8 : 0;
  const f16* p = rowbase + (size_t)r * (size_t)stride;
  v16h f;
  ((v8h*)&f)[0] = *(const v8h*)(p + kb);
  ((v8h*)&f)[1] = *(const v8h*)(p + kb + 16);
  return f;
}

// ---------------------------------------------------------------- utilities
__global__ void cvt_f32_to_f16_kernel(const float* __restrict__ src,
                                      f16* __restrict__ dst, int n) {
  int i = blockIdx.x * blockDim.x + threadIdx.x;
  if (i < n) dst[i] = (f16)src[i];
}

__global__ __launch_bounds__(256) void layernorm_f16_kernel(
    const float* __restrict__ x, const float* __restrict__ g,
    const float* __restrict__ b, f16* __restrict__ z) {
  const int D = D_MODEL;
  int row = blockIdx.x;
  int tid = threadIdx.x;
  const float* xr = x + (size_t)row * D;
  __shared__ float red[256];

  float s = 0.0f;
  for (int i = tid; i < D; i += 256) s += xr[i];
  red[tid] = s;
  __syncthreads();
  for (int w = 128; w > 0; w >>= 1) {
    if (tid < w) red[tid] += red[tid + w];
    __syncthreads();
  }
  float mean = red[0] / (float)D;
  __syncthreads();

  float vs = 0.0f;
  for (int i = tid; i < D; i += 256) {
    float d = xr[i] - mean;
    vs += d * d;
  }
  red[tid] = vs;
  __syncthreads();
  for (int w = 128; w > 0; w >>= 1) {
    if (tid < w) red[tid] += red[tid + w];
    __syncthreads();
  }
  float inv = rsqrtf(red[0] / (float)D + 1e-5f);

  f16* zr = z + (size_t)row * D;
  for (int i = tid; i < D; i += 256)
    zr[i] = (f16)((xr[i] - mean) * inv * g[i] + b[i]);
}

// ---------------------------------------------------------------- GEMM
// C[M,N] = A[M,K] @ W[N,K]^T + bias[N]
// EPI: 0 = store f16, 1 = exact-GELU then f16, 2 = += residual(f32), store f32
// Block: 256 thr = 8 waves; 128x128 tile; waves 4(M) x 2(N); wave = 32x64.
// K staged 64-deep, double-buffered via async LDS loads (8 b128 per thread
// per stage -> s_wait_asynccnt 8 overlaps the next prefetch with compute).
template <int EPI>
__global__ __launch_bounds__(256) void gemm_wt_kernel(
    const f16* __restrict__ A, const f16* __restrict__ W,
    const float* __restrict__ bias, const float* __restrict__ res,
    void* __restrict__ Cout, int M, int N, int K) {
  __shared__ f16 As[2][128][72];  // 72-half stride: 16B aligned, conflict-poor
  __shared__ f16 Ws[2][128][72];

  int tid = threadIdx.x, lane = tid & 31, wave = tid >> 5;
  int m0 = blockIdx.x * 128, n0 = blockIdx.y * 128;
  int wm = wave >> 1, wn = wave & 1;

  v8f acc[2][4];
#pragma unroll
  for (int s = 0; s < 2; ++s)
#pragma unroll
    for (int t = 0; t < 4; ++t) acc[s][t] = vzero8();

  // staging map: thread -> (row, 32-half segment); 4 x b128 each for A and W
  int srow = tid >> 1;
  int scol = (tid & 1) * 32;
  const f16* ga = A + (size_t)(m0 + srow) * K + scol;
  const f16* gw = W + (size_t)(n0 + srow) * K + scol;

  auto stage = [&](int k0, int buf) {
#pragma unroll
    for (int s = 0; s < 4; ++s)
      async_cp16(ga + k0 + s * 8, &As[buf][srow][scol + s * 8]);
#pragma unroll
    for (int s = 0; s < 4; ++s)
      async_cp16(gw + k0 + s * 8, &Ws[buf][srow][scol + s * 8]);
  };

  int nk = K >> 6;  // K chunks of 64
  stage(0, 0);
  int buf = 0;
  for (int it = 0; it < nk; ++it) {
    if (it + 1 < nk) {
      stage((it + 1) << 6, buf ^ 1);
      WAIT_ASYNC(8);  // current stage's 8 ops retired (in-order)
    } else {
      WAIT_ASYNC(0);
    }
    __syncthreads();

#pragma unroll
    for (int kc = 0; kc < 2; ++kc) {
      v16h af[2], bf[4];
#pragma unroll
      for (int s = 0; s < 2; ++s)
        af[s] = ld_frag(&As[buf][wm * 32 + s * 16][kc * 32], 72, lane);
#pragma unroll
      for (int t = 0; t < 4; ++t)
        bf[t] = ld_frag(&Ws[buf][wn * 64 + t * 16][kc * 32], 72, lane);
#pragma unroll
      for (int s = 0; s < 2; ++s)
#pragma unroll
        for (int t = 0; t < 4; ++t) acc[s][t] = wmma16(af[s], bf[t], acc[s][t]);
    }
    __syncthreads();
    buf ^= 1;
  }

  // C layout: VGPR e -> row e (lanes 0-15) / e+8 (lanes 16-31), lane&15 = col.
  int r = lane & 15;
  int mh = (lane >> 4) << 3;
#pragma unroll
  for (int s = 0; s < 2; ++s)
#pragma unroll
    for (int t = 0; t < 4; ++t)
#pragma unroll
      for (int e = 0; e < 8; ++e) {
        int gm = m0 + wm * 32 + s * 16 + e + mh;
        int gn = n0 + wn * 64 + t * 16 + r;
        float val = acc[s][t][e] + bias[gn];
        size_t idx = (size_t)gm * N + gn;
        if (EPI == 0) {
          ((f16*)Cout)[idx] = (f16)val;
        } else if (EPI == 1) {
          float gv = 0.5f * val * (1.0f + erff(val * 0.70710678118654752f));
          ((f16*)Cout)[idx] = (f16)gv;
        } else {
          ((float*)Cout)[idx] = val + res[idx];
        }
      }
}

// ---------------------------------------------------------------- attention
// Fused flash attention. Block = 8 waves sharing one (b,h) + 128 q rows;
// each wave owns a 16-row q strip. Key blocks of 64; K tile async-staged,
// V tile transposed on load so P@V B-fragments read contiguously.
__global__ __launch_bounds__(256) void attn_fused_kernel(
    const f16* __restrict__ q, const f16* __restrict__ kk,
    const f16* __restrict__ vv, const float* __restrict__ pbias,
    f16* __restrict__ o) {
  __shared__ f16 Kt[64][72];     // [key row][d]
  __shared__ f16 Vt[64][72];     // transposed: [d][key row]
  __shared__ f16 Pb[8][16][72];  // per-wave P staging (C-layout -> A-layout)

  int tid = threadIdx.x, lane = tid & 31, wave = tid >> 5;
  int bid = blockIdx.x;
  int qblk = bid & 7;
  int h = (bid >> 3) & (N_HEADS - 1);
  int b = bid >> 7;
  int q0 = qblk * 128 + wave * 16;

  int r = lane & 15;
  int kb = (lane & 16) ? 8 : 0;
  int mh = (lane >> 4) << 3;
  const float scale = 0.125f;  // 1/sqrt(64)

  // Q strip fragments (16 x 64) held in registers for the whole kernel.
  const f16* qrow = q + ((size_t)(b * SEQ + q0 + r)) * D_MODEL + h * DH;
  v16h qa0, qa1;
  ((v8h*)&qa0)[0] = *(const v8h*)(qrow + kb);
  ((v8h*)&qa0)[1] = *(const v8h*)(qrow + kb + 16);
  ((v8h*)&qa1)[0] = *(const v8h*)(qrow + 32 + kb);
  ((v8h*)&qa1)[1] = *(const v8h*)(qrow + 32 + kb + 16);

  v8f O[4];
#pragma unroll
  for (int t = 0; t < 4; ++t) O[t] = vzero8();
  float mrow[8], lrow[8];
#pragma unroll
  for (int e = 0; e < 8; ++e) { mrow[e] = -3.0e38f; lrow[e] = 0.0f; }

  int krow = tid >> 2, kseg = (tid & 3) * 16;  // K-tile staging map

  for (int k0 = 0; k0 < SEQ; k0 += 64) {
    __syncthreads();  // previous iteration's LDS reads done before overwrite
    {  // K tile: async copy, overlapped with the scalar V transpose below
      const f16* src =
          kk + ((size_t)(b * SEQ + k0 + krow)) * D_MODEL + h * DH + kseg;
      async_cp16(src, &Kt[krow][kseg]);
      async_cp16(src + 8, &Kt[krow][kseg + 8]);
    }
    // V tile, transposed on the way in: Vt[d][j] = V[k0+j][h*64+d]
    for (int idx = tid; idx < 64 * 64; idx += 256) {
      int jj = idx >> 6, dd = idx & 63;
      Vt[dd][jj] = vv[((size_t)(b * SEQ + k0 + jj)) * D_MODEL + h * DH + dd];
    }
    WAIT_ASYNC(0);
    __syncthreads();

    // S = scale * Q K^T + bias, four 16x16 tiles spanning 64 keys.
    v8f S[4];
#pragma unroll
    for (int t = 0; t < 4; ++t) {
      v16h kf0 = ld_frag(&Kt[t * 16][0], 72, lane);
      v16h kf1 = ld_frag(&Kt[t * 16][32], 72, lane);
      v8f s = wmma16(qa0, kf0, vzero8());
      s = wmma16(qa1, kf1, s);
#pragma unroll
      for (int e = 0; e < 8; ++e) {
        int gm = q0 + e + mh;
        int gk = k0 + t * 16 + r;
        s[e] = s[e] * scale + pbias[((size_t)(b * SEQ + gm)) * SEQ + gk];
      }
      S[t] = s;
    }

    // Online softmax; row stats replicated per 16-lane half, xor masks 1..8
    // never cross halves.
    float bm[8];
#pragma unroll
    for (int e = 0; e < 8; ++e)
      bm[e] = fmaxf(fmaxf(S[0][e], S[1][e]), fmaxf(S[2][e], S[3][e]));
    for (int m = 1; m <= 8; m <<= 1)
#pragma unroll
      for (int e = 0; e < 8; ++e) bm[e] = fmaxf(bm[e], __shfl_xor(bm[e], m, 32));
    float alpha[8];
#pragma unroll
    for (int e = 0; e < 8; ++e) {
      float mn = fmaxf(mrow[e], bm[e]);
      alpha[e] = __expf(mrow[e] - mn);
      mrow[e] = mn;
    }
    float bs[8];
#pragma unroll
    for (int e = 0; e < 8; ++e) bs[e] = 0.0f;
#pragma unroll
    for (int t = 0; t < 4; ++t)
#pragma unroll
      for (int e = 0; e < 8; ++e) {
        float p = __expf(S[t][e] - mrow[e]);
        S[t][e] = p;
        bs[e] += p;
      }
    for (int m = 1; m <= 8; m <<= 1)
#pragma unroll
      for (int e = 0; e < 8; ++e) bs[e] += __shfl_xor(bs[e], m, 32);
#pragma unroll
    for (int e = 0; e < 8; ++e) lrow[e] = lrow[e] * alpha[e] + bs[e];
#pragma unroll
    for (int t = 0; t < 4; ++t)
#pragma unroll
      for (int e = 0; e < 8; ++e) O[t][e] *= alpha[e];

    // Stage P (C layout) into per-wave LDS, reload as A fragments.
    // Same-wave DS ops are in-order: no barrier needed.
#pragma unroll
    for (int t = 0; t < 4; ++t)
#pragma unroll
      for (int e = 0; e < 8; ++e)
        Pb[wave][e + mh][t * 16 + r] = (f16)S[t][e];
    v16h pa0 = ld_frag(&Pb[wave][0][0], 72, lane);
    v16h pa1 = ld_frag(&Pb[wave][0][32], 72, lane);

    // O += P (16x64) @ V (64x64)
#pragma unroll
    for (int dt = 0; dt < 4; ++dt) {
      v16h vb0 = ld_frag(&Vt[dt * 16][0], 72, lane);
      v16h vb1 = ld_frag(&Vt[dt * 16][32], 72, lane);
      O[dt] = wmma16(pa0, vb0, O[dt]);
      O[dt] = wmma16(pa1, vb1, O[dt]);
    }
  }

#pragma unroll
  for (int dt = 0; dt < 4; ++dt)
#pragma unroll
    for (int e = 0; e < 8; ++e) {
      int gm = q0 + e + mh;
      o[((size_t)(b * SEQ + gm)) * D_MODEL + h * DH + dt * 16 + r] =
          (f16)(O[dt][e] / lrow[e]);
    }
}

// ---------------------------------------------------------------- launch
extern "C" void kernel_launch(void* const* d_in, const int* in_sizes, int n_in,
                              void* d_out, int out_size, void* d_ws,
                              size_t ws_size, hipStream_t stream) {
  const float* x      = (const float*)d_in[0];
  const float* pbias  = (const float*)d_in[1];
  const float* ln1_g  = (const float*)d_in[2];
  const float* ln1_b  = (const float*)d_in[3];
  const float* ln2_g  = (const float*)d_in[4];
  const float* ln2_b  = (const float*)d_in[5];
  const float* Wq     = (const float*)d_in[6];
  const float* Wk     = (const float*)d_in[7];
  const float* Wv     = (const float*)d_in[8];
  const float* bq     = (const float*)d_in[9];
  const float* bk     = (const float*)d_in[10];
  const float* bv     = (const float*)d_in[11];
  const float* Wo     = (const float*)d_in[12];
  const float* bo     = (const float*)d_in[13];
  const float* W1     = (const float*)d_in[14];
  const float* b1     = (const float*)d_in[15];
  const float* W2     = (const float*)d_in[16];
  const float* b2     = (const float*)d_in[17];

  char* ws = (char*)d_ws;
  size_t off = 0;
  auto wsalloc = [&](size_t bytes) -> void* {
    void* p = ws + off;
    off += (bytes + 255) & ~(size_t)255;
    return p;
  };

  const size_t DD = (size_t)D_MODEL * D_MODEL;
  const size_t DF = (size_t)D_MODEL * DIM_FF;
  const size_t MD = (size_t)MTOK * D_MODEL;
  const size_t MF = (size_t)MTOK * DIM_FF;

  f16* wq16  = (f16*)wsalloc(DD * 2);
  f16* wk16  = (f16*)wsalloc(DD * 2);
  f16* wv16  = (f16*)wsalloc(DD * 2);
  f16* wo16  = (f16*)wsalloc(DD * 2);
  f16* w116  = (f16*)wsalloc(DF * 2);
  f16* w216  = (f16*)wsalloc(DF * 2);
  f16* z16   = (f16*)wsalloc(MD * 2);
  f16* q16   = (f16*)wsalloc(MD * 2);
  f16* k16   = (f16*)wsalloc(MD * 2);
  f16* v16   = (f16*)wsalloc(MD * 2);
  f16* o16   = (f16*)wsalloc(MD * 2);
  f16* z216  = (f16*)wsalloc(MD * 2);
  f16* ff116 = (f16*)wsalloc(MF * 2);
  float* x1  = (float*)wsalloc(MD * 4);

  auto cvt = [&](const float* s, f16* d, size_t n) {
    cvt_f32_to_f16_kernel<<<(unsigned)((n + 255) / 256), 256, 0, stream>>>(
        s, d, (int)n);
  };
  cvt(Wq, wq16, DD);
  cvt(Wk, wk16, DD);
  cvt(Wv, wv16, DD);
  cvt(Wo, wo16, DD);
  cvt(W1, w116, DF);
  cvt(W2, w216, DF);

  // z = LN1(x)  (f16)
  layernorm_f16_kernel<<<MTOK, 256, 0, stream>>>(x, ln1_g, ln1_b, z16);

  dim3 gDD(MTOK / 128, D_MODEL / 128);
  dim3 gDF(MTOK / 128, DIM_FF / 128);

  // Q/K/V projections
  gemm_wt_kernel<0><<<gDD, 256, 0, stream>>>(z16, wq16, bq, nullptr, q16, MTOK,
                                             D_MODEL, D_MODEL);
  gemm_wt_kernel<0><<<gDD, 256, 0, stream>>>(z16, wk16, bk, nullptr, k16, MTOK,
                                             D_MODEL, D_MODEL);
  gemm_wt_kernel<0><<<gDD, 256, 0, stream>>>(z16, wv16, bv, nullptr, v16, MTOK,
                                             D_MODEL, D_MODEL);

  // fused biased softmax attention
  attn_fused_kernel<<<BSZ * N_HEADS * (SEQ / 128), 256, 0, stream>>>(
      q16, k16, v16, pbias, o16);

  // x1 = x + (o @ Wo^T + bo)
  gemm_wt_kernel<2><<<gDD, 256, 0, stream>>>(o16, wo16, bo, x, x1, MTOK,
                                             D_MODEL, D_MODEL);

  // z2 = LN2(x1)
  layernorm_f16_kernel<<<MTOK, 256, 0, stream>>>(x1, ln2_g, ln2_b, z216);

  // ff1 = gelu(z2 @ W1^T + b1)
  gemm_wt_kernel<1><<<gDF, 256, 0, stream>>>(z216, w116, b1, nullptr, ff116,
                                             MTOK, DIM_FF, D_MODEL);

  // out = x1 + (ff1 @ W2^T + b2)
  gemm_wt_kernel<2><<<gDD, 256, 0, stream>>>(ff116, w216, b2, x1, (float*)d_out,
                                             MTOK, D_MODEL, DIM_FF);
}
